// LCNNConvolution_71451075936922
// MI455X (gfx1250) — matrix operand order repaired and checked by reading, not compiled, not run on hardware
//
#include <hip/hip_runtime.h>

// LCNN convolution: out[r, :] = concat_{n=0..5}(X_sites[idx[r,n]]) @ W.T + b
//   rows r = site*6 + perm,  M = N_sites*6,  K = 384,  N = 64
// bf16 WMMA (f32 accumulate) on gfx1250, one 16-row tile per wave.

typedef __attribute__((ext_vector_type(16))) __bf16 v16bf;
typedef __attribute__((ext_vector_type(4)))  __bf16 v4bf;
typedef __attribute__((ext_vector_type(8)))  float  v8f;
typedef __attribute__((ext_vector_type(4)))  float  f4v;

#define FEAT 64
#define KDIM 384           // 6 neighbors * 64 features
#define LDSW 400           // padded LDS row stride (bf16 elems): 800B rows, 32B aligned

__global__ __launch_bounds__(256) void lcnn_conv_wmma(
    const float* __restrict__ X,      // (N_sites, 64) f32
    const int*   __restrict__ idx,    // (N_sites, 6, 6) int
    const float* __restrict__ W,      // (64, 384) f32 row-major
    const float* __restrict__ bias,   // (64,) f32
    float*       __restrict__ out,    // (N_sites, 6, 64) f32
    int n_rows, int n_tiles)
{
    __shared__ __bf16 Wlds[64 * LDSW];

    const int tid = threadIdx.x;
    // Stage W into LDS as bf16, float4 granularity: 6144 chunks / 256 threads = 24 each.
    // Independent iterations -> global_load_b128 clauses instead of serialized b32 loads.
#pragma unroll
    for (int j = 0; j < 24; ++j) {
        int i4 = tid + j * 256;            // chunk id; element offset = i4*4 = o*384 + k4*4
        int o  = i4 / 96;                  // 96 float4 per W row
        int k4 = i4 - o * 96;
        f4v w = *(const f4v*)(W + (size_t)i4 * 4);
        v4bf p;
        p[0] = (__bf16)w[0]; p[1] = (__bf16)w[1];
        p[2] = (__bf16)w[2]; p[3] = (__bf16)w[3];
        *(v4bf*)&Wlds[o * LDSW + k4 * 4] = p;   // 8B aligned ds_store_b64
    }
    __syncthreads();

    const int lane = tid & 31;
    const int m    = lane & 15;   // tile row (A) / output column within ct (B,C)
    const int kh   = lane >> 4;   // K-half selector per WMMA bf16 layout
    const int wave_global = blockIdx.x * 8 + (tid >> 5);
    const int n_waves     = gridDim.x * 8;

    for (int t = wave_global; t < n_tiles; t += n_waves) {   // wave-uniform: EXEC all-1s
        const int row_base = t * 16;
        int r  = row_base + m;
        int rs = r < n_rows ? r : (n_rows - 1);   // clamp gather for ragged tail

        // Preload this row's 6 neighbor indices.
        int nidx[6];
#pragma unroll
        for (int n = 0; n < 6; ++n) nidx[n] = idx[(size_t)rs * 6 + n];

        // Accumulators: lane covers one output column per ct -> init with bias (free add).
        v8f acc[4];
#pragma unroll
        for (int ct = 0; ct < 4; ++ct) {
            float bv = bias[ct * 16 + m];
#pragma unroll
            for (int e = 0; e < 8; ++e) acc[ct][e] = bv;
        }

#pragma unroll
        for (int kb = 0; kb < 12; ++kb) {          // K step = 32
            const int n     = kb >> 1;             // neighbor block (64 feats each)
            const int fbase = (kb & 1) * 32;       // offset within neighbor
            // A fragment, contiguous K order (same as B): kh=0 lanes take K 0..15, kh=1 K 16..31
            const float* src = X + (size_t)nidx[n] * FEAT + fbase + kh * 16;
            f4v f0 = *(const f4v*)(src);
            f4v f1 = *(const f4v*)(src + 4);
            f4v f2 = *(const f4v*)(src + 8);
            f4v f3 = *(const f4v*)(src + 12);
            v16bf a;
#pragma unroll
            for (int e = 0; e < 4; ++e) {
                a[e]      = (__bf16)f0[e];
                a[4 + e]  = (__bf16)f1[e];
                a[8 + e]  = (__bf16)f2[e];
                a[12 + e] = (__bf16)f3[e];
            }
            // B fragment: lane holds column ct*16+m, 16 consecutive K values
            // (kh=0 -> K 0..15, kh=1 -> K 16..31 of this k-step), VGPR r = {2r,2r+1}.
            const __bf16* wrow = &Wlds[m * LDSW + kb * 32 + kh * 16];
            const v16bf b0 = *(const v16bf*)(wrow);
            const v16bf b1 = *(const v16bf*)(wrow + 16 * LDSW);
            const v16bf b2 = *(const v16bf*)(wrow + 32 * LDSW);
            const v16bf b3 = *(const v16bf*)(wrow + 48 * LDSW);
            // Same A operand feeds 4 consecutive identical WMMAs: set the RA
            // operand-reuse hint on all but the last of the group (literal bools:
            // the builtin requires compile-time-constant reuse flags).
            acc[0] = __builtin_amdgcn_wmma_f32_16x16x32_bf16(
                false, a, false, b0, (short)0, acc[0], true, false);
            acc[1] = __builtin_amdgcn_wmma_f32_16x16x32_bf16(
                false, a, false, b1, (short)0, acc[1], true, false);
            acc[2] = __builtin_amdgcn_wmma_f32_16x16x32_bf16(
                false, a, false, b2, (short)0, acc[2], true, false);
            acc[3] = __builtin_amdgcn_wmma_f32_16x16x32_bf16(
                false, a, false, b3, (short)0, acc[3], false, false);
        }

        // C/D layout: VGPR v, lane l -> (row = v + 8*(l>=16), col = l&15).
        float* orow = out + (size_t)row_base * FEAT;
        if (row_base + 16 <= n_rows) {
            // Fast path: full tile, no guards (the only path when n_rows % 16 == 0).
#pragma unroll
            for (int ct = 0; ct < 4; ++ct)
#pragma unroll
                for (int v = 0; v < 8; ++v)
                    orow[(size_t)(v + kh * 8) * FEAT + ct * 16 + m] = acc[ct][v];
        } else {
#pragma unroll
            for (int ct = 0; ct < 4; ++ct)
#pragma unroll
                for (int v = 0; v < 8; ++v) {
                    int rr = v + kh * 8;
                    if (row_base + rr < n_rows)
                        orow[(size_t)rr * FEAT + ct * 16 + m] = acc[ct][v];
                }
        }
    }
}

extern "C" void kernel_launch(void* const* d_in, const int* in_sizes, int n_in,
                              void* d_out, int out_size, void* d_ws, size_t ws_size,
                              hipStream_t stream) {
    const float* X    = (const float*)d_in[0];   // X_sites (N,64) f32
    const int*   idx  = (const int*)d_in[1];     // X_NSs (N,6,6) int
    const float* W    = (const float*)d_in[3];   // (64,384) f32
    const float* bias = (const float*)d_in[4];   // (64,) f32
    float*       out  = (float*)d_out;           // (N,6,64) f32

    const int n_rows  = in_sizes[1] / 6;         // N_sites * N_PERM
    const int n_tiles = (n_rows + 15) / 16;
    int n_wg = (n_tiles + 7) / 8;
    if (n_wg > 2048) n_wg = 2048;                // grid-stride; 16384 waves max
    lcnn_conv_wmma<<<n_wg, 256, 0, stream>>>(X, idx, W, bias, out, n_rows, n_tiles);
}
